// AttnBlock_69355131896469
// MI455X (gfx1250) — matrix operand order repaired and checked by reading, not compiled, not run on hardware
//
#include <hip/hip_runtime.h>
#include <cstdint>

#ifndef __has_builtin
#define __has_builtin(x) 0
#endif

#if __has_builtin(__builtin_amdgcn_global_load_async_to_lds_b128)
#define ATTN_ASYNC_LDS 1
#else
#define ATTN_ASYNC_LDS 0
#endif

// ---------------------------------------------------------------- types
typedef __bf16 bf16;
typedef bf16 v16bf __attribute__((ext_vector_type(16)));
typedef float v8f  __attribute__((ext_vector_type(8)));

// async global->LDS builtin takes (AS1 int4*, AS3 int4*, int offset, int cpol)
typedef int v4i __attribute__((vector_size(16)));
typedef __attribute__((address_space(1))) v4i* gas1_v4i;
typedef __attribute__((address_space(3))) v4i* las3_v4i;

union FragB { v16bf v; uint4 q[2]; };
union FragC { v8f  v; float f[8]; };

__device__ __forceinline__ uint16_t f2bf(float f) {
    unsigned u = __float_as_uint(f);
    unsigned r = u + 0x7fffu + ((u >> 16) & 1u);   // round-to-nearest-even
    return (uint16_t)(r >> 16);
}

// A fragment (16x32 bf16, M x K): lane<16 -> row=lane, K = k0+{0..7} and k0+{16..23}
//                                 lane>=16 -> row=lane-16, K = k0+{8..15} and k0+{24..31}
__device__ __forceinline__ FragB load_fragA(const uint16_t* __restrict__ base, int ld,
                                            int row0, int k0, int lane) {
    FragB a;
    int r = lane & 15, h = lane >> 4;
    const uint16_t* p = base + (size_t)(row0 + r) * ld + k0 + h * 8;
    a.q[0] = *reinterpret_cast<const uint4*>(p);
    a.q[1] = *reinterpret_cast<const uint4*>(p + 16);
    return a;
}

// B fragment (32x16 bf16, K x N) loaded from Bt[n][k] row-major:
// lane n<16 -> col n, K = k0+{0..7},{16..23} ; lane n+16 -> col n, K = k0+{8..15},{24..31}
__device__ __forceinline__ FragB load_fragB(const uint16_t* __restrict__ base, int ld,
                                            int n0, int k0, int lane) {
    FragB b;
    int r = lane & 15, h = lane >> 4;
    const uint16_t* p = base + (size_t)(n0 + r) * ld + k0 + h * 8;
    b.q[0] = *reinterpret_cast<const uint4*>(p);
    b.q[1] = *reinterpret_cast<const uint4*>(p + 16);
    return b;
}

__device__ __forceinline__ v8f wmma_bf16(const FragB& a, const FragB& b, v8f c) {
    return __builtin_amdgcn_wmma_f32_16x16x32_bf16(false, a.v, false, b.v,
                                                   (short)0, c, false, false);
}

// ------------------------------------------------- GroupNorm statistics
__global__ void gn_stats_k(const float* __restrict__ x, float* __restrict__ stats) {
    __shared__ float ss[256], sq[256];
    int b = blockIdx.x >> 5;
    int g = blockIdx.x & 31;
    int t = threadIdx.x;
    const float* base = x + (size_t)b * 4096 * 512 + g * 16;
    float s = 0.f, s2 = 0.f;
    for (int p = t; p < 4096; p += 256) {
        const float4* pp = reinterpret_cast<const float4*>(base + (size_t)p * 512);
        #pragma unroll
        for (int q4 = 0; q4 < 4; ++q4) {
            float4 v = pp[q4];
            s  += v.x + v.y + v.z + v.w;
            s2 += v.x*v.x + v.y*v.y + v.z*v.z + v.w*v.w;
        }
    }
    ss[t] = s; sq[t] = s2;
    __syncthreads();
    for (int o = 128; o; o >>= 1) {
        if (t < o) { ss[t] += ss[t+o]; sq[t] += sq[t+o]; }
        __syncthreads();
    }
    if (t == 0) {
        float inv  = 1.0f / 65536.0f;
        float mean = ss[0] * inv;
        float var  = sq[0] * inv - mean * mean;
        stats[blockIdx.x*2]   = mean;
        stats[blockIdx.x*2+1] = rsqrtf(var + 1e-6f);
    }
}

// ------------------------------------------------- normalize + affine -> bf16
__global__ void gn_apply_k(const float* __restrict__ x, const float* __restrict__ stats,
                           const float* __restrict__ sc, const float* __restrict__ bi,
                           uint16_t* __restrict__ hn) {
    int idx = blockIdx.x * 256 + threadIdx.x;   // (b*4096+p)*32 + g
    int g  = idx & 31;
    int bp = idx >> 5;                          // b*4096 + p
    int sidx = ((bp >> 12) * 32 + g) * 2;
    float mean = stats[sidx];
    float rstd = stats[sidx + 1];
    const float* xp = x + (size_t)bp * 512 + g * 16;
    uint16_t outv[16];
    #pragma unroll
    for (int c = 0; c < 16; ++c) {
        float v = (xp[c] - mean) * rstd * sc[g*16 + c] + bi[g*16 + c];
        outv[c] = f2bf(v);
    }
    uint4* dst = reinterpret_cast<uint4*>(hn + (size_t)bp * 512 + g * 16);
    dst[0] = *reinterpret_cast<uint4*>(outv);
    dst[1] = *reinterpret_cast<uint4*>(outv + 8);
}

// ------------------------------------------------- weight transpose + bf16 cast
__global__ void wcvt_k(const float* __restrict__ w, uint16_t* __restrict__ wT) {
    int idx = blockIdx.x * 256 + threadIdx.x;   // 512*512
    int n = idx >> 9, k = idx & 511;
    wT[(size_t)n * 512 + k] = f2bf(w[(size_t)k * 512 + n]);   // wT[n][k] = w[k][n]
}

// ------------------------------------------------- projection GEMM (bf16 WMMA)
__global__ void proj_gemm_k(const uint16_t* __restrict__ A, const uint16_t* __restrict__ Bt,
                            const float* __restrict__ bias, uint16_t* __restrict__ out,
                            int transposeV) {
    int lane = threadIdx.x & 31;
    int gw = blockIdx.x * 8 + (threadIdx.x >> 5);
    int tn = gw & 31, tm = gw >> 5;
    int row0 = tm * 16, n0 = tn * 16;
    int r = lane & 15, h = lane >> 4;
    FragC acc;
    float bvv = bias[n0 + r];
    #pragma unroll
    for (int v = 0; v < 8; ++v) acc.f[v] = bvv;
    for (int kc = 0; kc < 16; ++kc) {
        FragB a = load_fragA(A, 512, row0, kc*32, lane);
        FragB b = load_fragB(Bt, 512, n0,  kc*32, lane);
        acc.v = wmma_bf16(a, b, acc.v);
    }
    #pragma unroll
    for (int v = 0; v < 8; ++v) {
        int row = row0 + v + 8*h;
        int col = n0 + r;
        uint16_t val = f2bf(acc.f[v]);
        if (transposeV) {                     // store vT[b][c][j]
            int bb = row >> 12, j = row & 4095;
            out[(size_t)bb * (512*4096) + (size_t)col * 4096 + j] = val;
        } else {
            out[(size_t)row * 512 + col] = val;
        }
    }
}

// ------------------------------------------------- flash attention
// block = (batch, 16-row q tile); 8 waves each own a 64-channel output slice.
// K tile (32 rows x 512, contiguous 32 KB) is staged once per block into LDS
// via async global->LDS (ASYNCcnt), score B-fragments come from ds_load_b128;
// online softmax; P bounced through LDS to become an A fragment for P.V WMMAs.
__global__ void attn_k(const uint16_t* __restrict__ q, const uint16_t* __restrict__ kmat,
                       const uint16_t* __restrict__ vT, uint16_t* __restrict__ aout) {
    __shared__ __align__(16) uint16_t Kt[32 * 512];     // 32 KB K tile
    __shared__ __align__(16) uint16_t Pt[8][16 * 32];   // 8 KB per-wave P tiles
    int lane = threadIdx.x & 31;
    int wid  = threadIdx.x >> 5;
    int t    = threadIdx.x;
    int it = blockIdx.x & 255;
    int b  = blockIdx.x >> 8;
    int i0 = it * 16;
    int c0 = wid * 64;
    int r = lane & 15, h = lane >> 4;

    const uint16_t* qb = q    + (size_t)b * 4096 * 512;
    const uint16_t* kb = kmat + (size_t)b * 4096 * 512;
    const uint16_t* vb = vT   + (size_t)b * 512 * 4096;

    FragB qa[16];                                   // q tile resident in VGPRs
    #pragma unroll
    for (int kc = 0; kc < 16; ++kc) qa[kc] = load_fragA(qb, 512, i0, kc*32, lane);

    FragC o[4];
    float m[8], l[8];
    #pragma unroll
    for (int v = 0; v < 8; ++v) { m[v] = -1e30f; l[v] = 0.f; }
    #pragma unroll
    for (int n = 0; n < 4; ++n)
        #pragma unroll
        for (int v = 0; v < 8; ++v) o[n].f[v] = 0.f;
    const float scale = 0.044194173824159216f;      // 512^-0.5

    for (int j0 = 0; j0 < 4096; j0 += 32) {
        // ---- stage K rows j0..j0+31 into LDS (one contiguous 32 KB chunk)
#if ATTN_ASYNC_LDS
        {
            const char* gsrc = (const char*)(kb + (size_t)j0 * 512);
            char* lbase = (char*)Kt;
            #pragma unroll
            for (int i = 0; i < 8; ++i) {
                unsigned off = (unsigned)(i * 256 + t) * 16u;
                __builtin_amdgcn_global_load_async_to_lds_b128(
                    (gas1_v4i)(uintptr_t)(gsrc + off),
                    (las3_v4i)(unsigned)(uintptr_t)(lbase + off),
                    0, 0);
            }
#if __has_builtin(__builtin_amdgcn_s_wait_asynccnt)
            __builtin_amdgcn_s_wait_asynccnt(0);
#else
            asm volatile("s_wait_asynccnt 0x0" ::: "memory");
#endif
        }
#else
        {
            const uint4* gsrc = reinterpret_cast<const uint4*>(kb + (size_t)j0 * 512);
            uint4* ldst = reinterpret_cast<uint4*>(Kt);
            #pragma unroll
            for (int i = 0; i < 8; ++i) ldst[i * 256 + t] = gsrc[i * 256 + t];
        }
#endif
        __syncthreads();
        if (j0 + 32 < 4096) __builtin_prefetch(kb + (size_t)(j0 + 32) * 512, 0, 0);

        // ---- S = q . k^T  (B fragments out of LDS)
        FragC s0, s1;
        #pragma unroll
        for (int v = 0; v < 8; ++v) { s0.f[v] = 0.f; s1.f[v] = 0.f; }
        for (int kc = 0; kc < 16; ++kc) {
            FragB b0, b1;
            {
                const uint16_t* p0 = &Kt[(size_t)r * 512 + kc*32 + h*8];
                b0.q[0] = *reinterpret_cast<const uint4*>(p0);
                b0.q[1] = *reinterpret_cast<const uint4*>(p0 + 16);
                const uint16_t* p1 = &Kt[(size_t)(16 + r) * 512 + kc*32 + h*8];
                b1.q[0] = *reinterpret_cast<const uint4*>(p1);
                b1.q[1] = *reinterpret_cast<const uint4*>(p1 + 16);
            }
            s0.v = wmma_bf16(qa[kc], b0, s0.v);
            s1.v = wmma_bf16(qa[kc], b1, s1.v);
        }

        // ---- online softmax per row (stats replicated across the 16-lane half)
        #pragma unroll
        for (int v = 0; v < 8; ++v) {
            float a0 = s0.f[v]*scale, a1 = s1.f[v]*scale;
            float tmx = fmaxf(a0, a1);
            for (int msk = 8; msk; msk >>= 1) tmx = fmaxf(tmx, __shfl_xor(tmx, msk, 16));
            float mn = fmaxf(m[v], tmx);
            float alpha = __expf(m[v] - mn);
            m[v] = mn;
            float p0 = __expf(a0 - mn), p1 = __expf(a1 - mn);
            float rs = p0 + p1;
            for (int msk = 8; msk; msk >>= 1) rs += __shfl_xor(rs, msk, 16);
            l[v] = l[v]*alpha + rs;
            #pragma unroll
            for (int n = 0; n < 4; ++n) o[n].f[v] *= alpha;
            int row = v + 8*h;                      // C-frag -> LDS (row-major 16x32)
            Pt[wid][row*32 + r]      = f2bf(p0);
            Pt[wid][row*32 + 16 + r] = f2bf(p1);
        }
        __syncthreads();
        FragB pa;                                   // reload P as A fragment (16x32)
        {
            const uint16_t* sp = &Pt[wid][r*32 + h*8];
            pa.q[0] = *reinterpret_cast<const uint4*>(sp);
            pa.q[1] = *reinterpret_cast<const uint4*>(sp + 16);
        }

        // ---- O += P . V  for this wave's 64-channel slice
        #pragma unroll
        for (int n = 0; n < 4; ++n) {
            FragB bv = load_fragB(vb, 4096, c0 + n*16, j0, lane);
            o[n].v = wmma_bf16(pa, bv, o[n].v);
        }
        __syncthreads();   // protect Kt + Pt before next iteration's staging
    }

    #pragma unroll
    for (int v = 0; v < 8; ++v) {
        int row = i0 + v + 8*h;
        float invl = 1.0f / l[v];
        size_t base = ((size_t)b*4096 + row) * 512;
        #pragma unroll
        for (int n = 0; n < 4; ++n)
            aout[base + c0 + n*16 + r] = f2bf(o[n].f[v] * invl);
    }
}

// ------------------------------------------------- output projection + residual
__global__ void outproj_k(const uint16_t* __restrict__ A, const uint16_t* __restrict__ Bt,
                          const float* __restrict__ bias, const float* __restrict__ x,
                          float* __restrict__ y) {
    int lane = threadIdx.x & 31;
    int gw = blockIdx.x * 8 + (threadIdx.x >> 5);
    int tn = gw & 31, tm = gw >> 5;
    int row0 = tm * 16, n0 = tn * 16;
    int r = lane & 15, h = lane >> 4;
    FragC acc;
    float bvv = bias[n0 + r];
    #pragma unroll
    for (int v = 0; v < 8; ++v) acc.f[v] = bvv;
    for (int kc = 0; kc < 16; ++kc) {
        FragB a = load_fragA(A, 512, row0, kc*32, lane);
        FragB b = load_fragB(Bt, 512, n0,  kc*32, lane);
        acc.v = wmma_bf16(a, b, acc.v);
    }
    #pragma unroll
    for (int v = 0; v < 8; ++v) {
        size_t idx = (size_t)(row0 + v + 8*h) * 512 + n0 + r;
        y[idx] = x[idx] + acc.f[v];
    }
}

// ------------------------------------------------- host launcher
extern "C" void kernel_launch(void* const* d_in, const int* in_sizes, int n_in,
                              void* d_out, int out_size, void* d_ws, size_t ws_size,
                              hipStream_t stream) {
    (void)in_sizes; (void)n_in; (void)out_size; (void)ws_size;
    const float* x  = (const float*)d_in[0];
    const float* gs = (const float*)d_in[1];
    const float* gb = (const float*)d_in[2];
    const float* wq = (const float*)d_in[3];
    const float* bq = (const float*)d_in[4];
    const float* wk = (const float*)d_in[5];
    const float* bk = (const float*)d_in[6];
    const float* wv = (const float*)d_in[7];
    const float* bv = (const float*)d_in[8];
    const float* wo = (const float*)d_in[9];
    const float* bo = (const float*)d_in[10];
    float* y = (float*)d_out;

    char* ws = (char*)d_ws;
    size_t off = 0;
    auto carve = [&](size_t bytes) {
        char* p = ws + off;
        off = (off + bytes + 255) & ~(size_t)255;
        return p;
    };
    const size_t NTOK = 4 * 4096;
    uint16_t* hn    = (uint16_t*)carve(NTOK * 512 * 2);
    uint16_t* wqT   = (uint16_t*)carve(512 * 512 * 2);
    uint16_t* wkT   = (uint16_t*)carve(512 * 512 * 2);
    uint16_t* wvT   = (uint16_t*)carve(512 * 512 * 2);
    uint16_t* woT   = (uint16_t*)carve(512 * 512 * 2);
    uint16_t* qb    = (uint16_t*)carve(NTOK * 512 * 2);
    uint16_t* kb    = (uint16_t*)carve(NTOK * 512 * 2);
    uint16_t* vTb   = (uint16_t*)carve(NTOK * 512 * 2);
    uint16_t* aoutb = (uint16_t*)carve(NTOK * 512 * 2);
    float*    stats = (float*)carve(4 * 32 * 2 * sizeof(float));

    gn_stats_k<<<128, 256, 0, stream>>>(x, stats);
    gn_apply_k<<<(4 * 4096 * 32) / 256, 256, 0, stream>>>(x, stats, gs, gb, hn);
    wcvt_k<<<1024, 256, 0, stream>>>(wq, wqT);
    wcvt_k<<<1024, 256, 0, stream>>>(wk, wkT);
    wcvt_k<<<1024, 256, 0, stream>>>(wv, wvT);
    wcvt_k<<<1024, 256, 0, stream>>>(wo, woT);
    proj_gemm_k<<<4096, 256, 0, stream>>>(hn, wqT, bq, qb, 0);
    proj_gemm_k<<<4096, 256, 0, stream>>>(hn, wkT, bk, kb, 0);
    proj_gemm_k<<<4096, 256, 0, stream>>>(hn, wvT, bv, vTb, 1);
    attn_k<<<1024, 256, 0, stream>>>(qb, kb, vTb, aoutb);
    outproj_k<<<4096, 256, 0, stream>>>(aoutb, woT, bo, x, y);
}
